// NonParallelProsodyPredictor_48868137894145
// MI455X (gfx1250) — matrix operand order repaired and checked by reading, not compiled, not run on hardware
//
#include <hip/hip_runtime.h>
#include <hip/hip_bf16.h>
#include <stdint.h>

// Problem dims (match reference)
#define Bsz 64
#define Ssz 512
#define Dsz 512
#define Hsz 1024          // GRU hidden = 2*D
#define Ktap 5
#define G3  3072          // 3*H
#define NM  6144          // merged serial-GEMM N: [W_ih[:,D:] ; W_hh]
#define EPSV 1e-5f

typedef __attribute__((ext_vector_type(16))) _Float16 v16h;
typedef __attribute__((ext_vector_type(8)))  _Float16 v8h;
typedef __attribute__((ext_vector_type(8)))  float    v8f;

// ---------------------------------------------------------------------------
// WMMA helpers (CDNA5 16x16x32 f16 -> f32). Layouts per CDNA5 ISA:
//   A (16x32, MxK): lane m=l&15, ksel=l>>4; VGPR v<4 -> K=2v+8*ksel,
//                   v>=4 -> K=2v+8+8*ksel  == two contiguous 8-half runs.
//   B (32x16, KxN): lane n=l&15, ksel=l>>4; VGPR v -> same K as A
//                   == two contiguous 8-half runs when packed [N][K].
//   C/D (16x16):    lane n=l&15, mh=l>>4; VGPR r -> row r+8*mh, col n.
// ---------------------------------------------------------------------------
__device__ __forceinline__ v8f vzero8() {
  v8f z;
  #pragma unroll
  for (int i = 0; i < 8; ++i) z[i] = 0.0f;
  return z;
}

__device__ __forceinline__ v8f wmma_f16(v16h a, v16h b, v8f c) {
  return __builtin_amdgcn_wmma_f32_16x16x32_f16(
      /*neg_a=*/false, a, /*neg_b=*/false, b,
      /*c_mod=*/(short)0, c, /*reuse_a=*/false, /*reuse_b=*/false);
}

// A fragment: row-major A, element (m,kk) at base[m*ldm + kk].
__device__ __forceinline__ v16h load_a16(const _Float16* base, int ldm, int lane) {
  const int m = lane & 15, ks = lane >> 4;
  const _Float16* q = base + (size_t)m * ldm + ks * 8;
  v8h lo = *(const v8h*)q;          // K = ks*8 .. ks*8+7
  v8h hi = *(const v8h*)(q + 16);   // K = 16+ks*8 .. 16+ks*8+7
  v16h a;
  #pragma unroll
  for (int i = 0; i < 8; ++i) { a[i] = lo[i]; a[8 + i] = hi[i]; }
  return a;
}

// B fragment from weights packed column-major-ish: PB[n*ldk + k].
__device__ __forceinline__ v16h load_b16(const _Float16* pk, int ldk,
                                         int n0, int kc0, int lane) {
  const int n = n0 + (lane & 15), ks = lane >> 4;
  const _Float16* q = pk + (size_t)n * ldk + kc0 + ks * 8;
  v8h lo = *(const v8h*)q;
  v8h hi = *(const v8h*)(q + 16);
  v16h b;
  #pragma unroll
  for (int i = 0; i < 8; ++i) { b[i] = lo[i]; b[8 + i] = hi[i]; }
  return b;
}

__device__ __forceinline__ float sigmoidf_(float x) {
  return 1.0f / (1.0f + __expf(-x));
}

// ---------------------------------------------------------------------------
// Weight repack kernels (f32 -> f16, GEMM-friendly layouts). Run once/launch.
// ---------------------------------------------------------------------------
__global__ void pack_conv_kernel(const float* __restrict__ w,      // [Dout][Din][K]
                                 _Float16* __restrict__ wp) {      // [Dout][K][Din] as B[n][ktot]
  int i = blockIdx.x * 256 + threadIdx.x;
  if (i >= Dsz * Dsz * Ktap) return;
  int k = i % Ktap; int t = i / Ktap; int din = t % Dsz; int dout = t / Dsz;
  wp[(size_t)dout * (Ktap * Dsz) + k * Dsz + din] = (_Float16)w[i];
}

__global__ void pack_wx_kernel(const float* __restrict__ W_ih,     // [3H][D+H]
                               _Float16* __restrict__ Wxp) {       // [3H][D] as B[n][k]
  int i = blockIdx.x * 256 + threadIdx.x;
  if (i >= G3 * Dsz) return;
  int din = i % Dsz; int dout = i / Dsz;
  Wxp[i] = (_Float16)W_ih[(size_t)dout * (Dsz + Hsz) + din];
}

__global__ void pack_whm_kernel(const float* __restrict__ W_ih,    // [3H][D+H]
                                const float* __restrict__ W_hh,    // [3H][H]
                                _Float16* __restrict__ Whm) {      // [6144][1024] as B[n][k]
  int i = blockIdx.x * 256 + threadIdx.x;
  if (i >= NM * Hsz) return;
  int k = i % Hsz; int n = i / Hsz;
  float v = (n < G3) ? W_ih[(size_t)n * (Dsz + Hsz) + Dsz + k]
                     : W_hh[(size_t)(n - G3) * Hsz + k];
  Whm[i] = (_Float16)v;
}

// ---------------------------------------------------------------------------
// Conv1d(K=5, SAME) + bias + LayerNorm(channels) + ReLU + mask -> f16 out.
// One workgroup = 16 tokens of one batch row; 8 waves x 4 N-tiles = 512 chans.
// ---------------------------------------------------------------------------
template <typename TIN>
__global__ void __launch_bounds__(256) conv_ln_relu_kernel(
    const TIN* __restrict__ xin, const float* __restrict__ bias,
    const float* __restrict__ gamma, const float* __restrict__ beta,
    const uint8_t* __restrict__ mask, const _Float16* __restrict__ Wp,
    _Float16* __restrict__ xout) {
  __shared__ _Float16 Alds[20 * Dsz];   // 20KB: token window [t0-2, t0+17]
  __shared__ float    Olds[16 * Dsz];   // 32KB: conv output for LN epilogue

  const int t0 = blockIdx.x * 16;
  const int b  = blockIdx.y;
  const int tid = threadIdx.x;

  for (int i = tid; i < 20 * Dsz; i += 256) {
    int r = i >> 9, c = i & (Dsz - 1);
    int t = t0 + r - 2;
    float v = 0.0f;
    if (t >= 0 && t < Ssz) v = (float)xin[((size_t)b * Ssz + t) * Dsz + c];
    Alds[i] = (_Float16)v;
  }
  __syncthreads();

  const int wave = tid >> 5, lane = tid & 31;
  v8f acc[4];
  #pragma unroll
  for (int i = 0; i < 4; ++i) acc[i] = vzero8();

  for (int k = 0; k < Ktap; ++k) {
    for (int kc = 0; kc < Dsz / 32; ++kc) {
      v16h a = load_a16(Alds + k * Dsz + kc * 32, Dsz, lane);
      #pragma unroll
      for (int i = 0; i < 4; ++i) {
        v16h bf = load_b16(Wp, Ktap * Dsz, (wave * 4 + i) * 16,
                           k * Dsz + kc * 32, lane);
        acc[i] = wmma_f16(a, bf, acc[i]);
      }
    }
  }

  { // write conv+bias to LDS for the LayerNorm epilogue
    const int n = lane & 15, mh = lane >> 4;
    #pragma unroll
    for (int i = 0; i < 4; ++i) {
      const int col = (wave * 4 + i) * 16 + n;
      const float bi = bias[col];
      #pragma unroll
      for (int r = 0; r < 8; ++r)
        Olds[(r + 8 * mh) * Dsz + col] = acc[i][r] + bi;
    }
  }
  __syncthreads();

  // LayerNorm over 512 channels, 2 tokens per wave
  for (int tk = wave * 2; tk < wave * 2 + 2; ++tk) {
    float s = 0.0f, ss = 0.0f;
    for (int c = lane; c < Dsz; c += 32) {
      float v = Olds[tk * Dsz + c];
      s += v; ss += v * v;
    }
    #pragma unroll
    for (int off = 16; off >= 1; off >>= 1) {
      s  += __shfl_xor(s, off, 32);
      ss += __shfl_xor(ss, off, 32);
    }
    const float m   = s * (1.0f / Dsz);
    const float var = ss * (1.0f / Dsz) - m * m;
    const float inv = rsqrtf(var + EPSV);
    const int t = t0 + tk;
    const bool msk = mask[(size_t)b * Ssz + t] != 0;
    for (int c = lane; c < Dsz; c += 32) {
      float y = (Olds[tk * Dsz + c] - m) * inv * gamma[c] + beta[c];
      y = fmaxf(y, 0.0f);
      if (msk) y = 0.0f;
      xout[((size_t)b * Ssz + t) * Dsz + c] = (_Float16)y;
    }
  }
}

// ---------------------------------------------------------------------------
// Xproj = x2 @ W_ih[:, :D]^T + b_ih  for all 32768 tokens (parallel WMMA).
// grid = (tokens/16, 3072/512); 8 waves x 4 N-tiles each.
// ---------------------------------------------------------------------------
__global__ void __launch_bounds__(256) xproj_kernel(
    const _Float16* __restrict__ x2, const _Float16* __restrict__ Wxp,
    const float* __restrict__ b_ih, float* __restrict__ Xproj) {
  __shared__ _Float16 Alds[16 * Dsz];   // 16KB
  const int m0 = blockIdx.x * 16;
  const int nb = blockIdx.y * 512;
  const int tid = threadIdx.x;

  for (int i = tid; i < 16 * Dsz; i += 256)
    Alds[i] = x2[((size_t)(m0 + (i >> 9))) * Dsz + (i & (Dsz - 1))];
  __syncthreads();

  const int wave = tid >> 5, lane = tid & 31;
  v8f acc[4];
  #pragma unroll
  for (int i = 0; i < 4; ++i) acc[i] = vzero8();

  for (int kc = 0; kc < Dsz / 32; ++kc) {
    v16h a = load_a16(Alds + kc * 32, Dsz, lane);
    #pragma unroll
    for (int i = 0; i < 4; ++i) {
      v16h bf = load_b16(Wxp, Dsz, nb + (wave * 4 + i) * 16, kc * 32, lane);
      acc[i] = wmma_f16(a, bf, acc[i]);
    }
  }
  const int n = lane & 15, mh = lane >> 4;
  #pragma unroll
  for (int i = 0; i < 4; ++i) {
    const int col = nb + (wave * 4 + i) * 16 + n;
    const float bi = b_ih[col];
    #pragma unroll
    for (int r = 0; r < 8; ++r)
      Xproj[((size_t)(m0 + r + 8 * mh)) * G3 + col] = acc[i][r] + bi;
  }
}

// ---------------------------------------------------------------------------
// Persistent GRU: 64 workgroups, grid barrier per time step.
// WG g owns hidden cols [g*16, g*16+16); per step computes its 6 gate column
// groups (u_r,u_z,u_n from W_ih h-part; v_r,v_z,v_n from W_hh) for all 64
// batches via WMMA (M=64 -> 4 M-tiles; 8 waves, each: 1 M-tile x 3 N-groups).
// ---------------------------------------------------------------------------
__device__ __forceinline__ void grid_barrier(unsigned* bar, unsigned expected) {
  __syncthreads();
  if (threadIdx.x == 0) {
    __threadfence();
    atomicAdd(bar, 1u);
    while (__hip_atomic_load(bar, __ATOMIC_ACQUIRE, __HIP_MEMORY_SCOPE_AGENT) <
           expected)
      __builtin_amdgcn_s_sleep(2);
  }
  __syncthreads();
}

__global__ void __launch_bounds__(256, 1) gru_kernel(
    const float* __restrict__ Xproj, const _Float16* __restrict__ Whm,
    const float* __restrict__ b_hh, const uint8_t* __restrict__ mask,
    _Float16* __restrict__ h16, float* __restrict__ h32,
    float* __restrict__ outputs, unsigned* __restrict__ bar) {
  __shared__ float glds[4 * 6 * 256];   // 24KB gate tiles

  const int g = blockIdx.x;             // 0..63
  const int tid = threadIdx.x;
  const int wave = tid >> 5, lane = tid & 31;
  const int mtile = wave >> 1;          // 0..3  (batch tile)
  const int ngset = wave & 1;           // 0: u gates, 1: v gates

  // h0 = 0 (buffer 0); each WG zeroes its own hidden slice.
  #pragma unroll
  for (int it = 0; it < 4; ++it) {
    const int idx = tid + it * 256;
    const int bb = idx >> 4, jl = idx & 15;
    const int j = g * 16 + jl;
    h32[bb * Hsz + j] = 0.0f;
    h16[bb * Hsz + j] = (_Float16)0.0f;
  }
  grid_barrier(bar, 64u * 1u);

  for (int t = 0; t < Ssz; ++t) {
    const int cur = t & 1, nxt = cur ^ 1;
    const _Float16* hp16 = h16 + (size_t)cur * Bsz * Hsz;
    const float*    hp32 = h32 + (size_t)cur * Bsz * Hsz;
    _Float16* hn16 = h16 + (size_t)nxt * Bsz * Hsz;
    float*    hn32 = h32 + (size_t)nxt * Bsz * Hsz;

    v8f a0 = vzero8(), a1 = vzero8(), a2 = vzero8();
    for (int kc = 0; kc < Hsz / 32; ++kc) {
      v16h a = load_a16(hp16 + (size_t)mtile * 16 * Hsz + kc * 32, Hsz, lane);
      v16h b0 = load_b16(Whm, Hsz, (ngset * 3 + 0) * Hsz + g * 16, kc * 32, lane);
      a0 = wmma_f16(a, b0, a0);
      v16h b1 = load_b16(Whm, Hsz, (ngset * 3 + 1) * Hsz + g * 16, kc * 32, lane);
      a1 = wmma_f16(a, b1, a1);
      v16h b2 = load_b16(Whm, Hsz, (ngset * 3 + 2) * Hsz + g * 16, kc * 32, lane);
      a2 = wmma_f16(a, b2, a2);
    }
    { // spill gate tiles to LDS for cross-wave gate fusion
      const int n = lane & 15, mh = lane >> 4;
      #pragma unroll
      for (int r = 0; r < 8; ++r) {
        const int row = (r + 8 * mh) * 16 + n;
        glds[(mtile * 6 + ngset * 3 + 0) * 256 + row] = a0[r];
        glds[(mtile * 6 + ngset * 3 + 1) * 256 + row] = a1[r];
        glds[(mtile * 6 + ngset * 3 + 2) * 256 + row] = a2[r];
      }
    }
    __syncthreads();

    // Gate fusion + state update for this WG's 64x16 (batch x hidden) slice.
    #pragma unroll
    for (int it = 0; it < 4; ++it) {
      const int idx = tid + it * 256;
      const int bb = idx >> 4, jl = idx & 15;
      const int j = g * 16 + jl;
      const int mt = bb >> 4, mr = bb & 15;
      const float ur = glds[(mt * 6 + 0) * 256 + mr * 16 + jl];
      const float uz = glds[(mt * 6 + 1) * 256 + mr * 16 + jl];
      const float un = glds[(mt * 6 + 2) * 256 + mr * 16 + jl];
      const float vr = glds[(mt * 6 + 3) * 256 + mr * 16 + jl];
      const float vz = glds[(mt * 6 + 4) * 256 + mr * 16 + jl];
      const float vn = glds[(mt * 6 + 5) * 256 + mr * 16 + jl];
      const size_t tok = (size_t)bb * Ssz + t;
      const float* xp = Xproj + tok * G3;   // b_ih already folded in
      const float hprev = hp32[bb * Hsz + j];
      const float r = sigmoidf_(xp[j] + ur + vr + b_hh[j]);
      const float z = sigmoidf_(xp[Hsz + j] + uz + vz + b_hh[Hsz + j]);
      const float nn = tanhf(xp[2 * Hsz + j] + un + r * (vn + b_hh[2 * Hsz + j]));
      const float hnew = (1.0f - z) * nn + z * hprev;
      hn32[bb * Hsz + j] = hnew;
      hn16[bb * Hsz + j] = (_Float16)hnew;
      outputs[tok * Hsz + j] = (mask[tok] != 0) ? 0.0f : hnew;
    }
    grid_barrier(bar, 64u * (unsigned)(t + 2));
  }
}

// ---------------------------------------------------------------------------
// Bottleneck: out[tok, c] = sum_j outputs[tok, j] * W_bn[c, j] + b_bn[c].
// One wave per token (memory-bound sweep of the 128MB outputs buffer).
// ---------------------------------------------------------------------------
__global__ void __launch_bounds__(256) bottleneck_kernel(
    const float* __restrict__ outputs, const float* __restrict__ W_bn,
    const float* __restrict__ b_bn, float* __restrict__ out) {
  const int wave = threadIdx.x >> 5, lane = threadIdx.x & 31;
  const size_t tok = (size_t)blockIdx.x * 8 + wave;
  float s0 = 0.f, s1 = 0.f, s2 = 0.f, s3 = 0.f;
  for (int j = lane; j < Hsz; j += 32) {
    const float o = outputs[tok * Hsz + j];
    s0 += o * W_bn[0 * Hsz + j];
    s1 += o * W_bn[1 * Hsz + j];
    s2 += o * W_bn[2 * Hsz + j];
    s3 += o * W_bn[3 * Hsz + j];
  }
  #pragma unroll
  for (int off = 16; off >= 1; off >>= 1) {
    s0 += __shfl_xor(s0, off, 32);
    s1 += __shfl_xor(s1, off, 32);
    s2 += __shfl_xor(s2, off, 32);
    s3 += __shfl_xor(s3, off, 32);
  }
  if (lane == 0) {
    out[tok * 4 + 0] = s0 + b_bn[0];
    out[tok * 4 + 1] = s1 + b_bn[1];
    out[tok * 4 + 2] = s2 + b_bn[2];
    out[tok * 4 + 3] = s3 + b_bn[3];
  }
}

// ---------------------------------------------------------------------------
// kernel_launch
// Workspace budget (~620MB): bar | x1 f16 32MB | x2 f16 32MB | packed weights
// ~21MB | Xproj f32 403MB | outputs f32 134MB | h32/h16 double buffers.
// ---------------------------------------------------------------------------
extern "C" void kernel_launch(void* const* d_in, const int* in_sizes, int n_in,
                              void* d_out, int out_size, void* d_ws,
                              size_t ws_size, hipStream_t stream) {
  const float*   h_text  = (const float*)d_in[0];
  const uint8_t* mask    = (const uint8_t*)d_in[1];
  const float*   conv_w1 = (const float*)d_in[2];
  const float*   conv_b1 = (const float*)d_in[3];
  const float*   ln_g1   = (const float*)d_in[4];
  const float*   ln_b1   = (const float*)d_in[5];
  const float*   conv_w2 = (const float*)d_in[6];
  const float*   conv_b2 = (const float*)d_in[7];
  const float*   ln_g2   = (const float*)d_in[8];
  const float*   ln_b2   = (const float*)d_in[9];
  const float*   W_ih    = (const float*)d_in[10];
  const float*   W_hh    = (const float*)d_in[11];
  const float*   b_ih    = (const float*)d_in[12];
  const float*   b_hh    = (const float*)d_in[13];
  const float*   W_bn    = (const float*)d_in[14];
  const float*   b_bn    = (const float*)d_in[15];
  float* out = (float*)d_out;
  (void)in_sizes; (void)n_in; (void)out_size; (void)ws_size;

  uintptr_t p = (uintptr_t)d_ws;
  auto alloc = [&](size_t bytes) -> void* {
    void* r = (void*)p;
    p += (bytes + 255) & ~(size_t)255;
    return r;
  };
  unsigned*  bar   = (unsigned*)alloc(256);
  _Float16*  x1    = (_Float16*)alloc((size_t)Bsz * Ssz * Dsz * 2);
  _Float16*  x2    = (_Float16*)alloc((size_t)Bsz * Ssz * Dsz * 2);
  _Float16*  Wc1   = (_Float16*)alloc((size_t)Ktap * Dsz * Dsz * 2);
  _Float16*  Wc2   = (_Float16*)alloc((size_t)Ktap * Dsz * Dsz * 2);
  _Float16*  Wxp   = (_Float16*)alloc((size_t)G3 * Dsz * 2);
  _Float16*  Whm   = (_Float16*)alloc((size_t)NM * Hsz * 2);
  float*     Xproj = (float*)alloc((size_t)Bsz * Ssz * G3 * 4);
  float*     outs  = (float*)alloc((size_t)Bsz * Ssz * Hsz * 4);
  float*     h32   = (float*)alloc((size_t)2 * Bsz * Hsz * 4);
  _Float16*  h16   = (_Float16*)alloc((size_t)2 * Bsz * Hsz * 2);

  (void)hipMemsetAsync(bar, 0, 256, stream);

  // Weight repack (f32 -> f16, B-fragment-friendly [n][k] layouts)
  pack_conv_kernel<<<(Dsz * Dsz * Ktap + 255) / 256, 256, 0, stream>>>(conv_w1, Wc1);
  pack_conv_kernel<<<(Dsz * Dsz * Ktap + 255) / 256, 256, 0, stream>>>(conv_w2, Wc2);
  pack_wx_kernel<<<(G3 * Dsz + 255) / 256, 256, 0, stream>>>(W_ih, Wxp);
  pack_whm_kernel<<<(NM * Hsz + 255) / 256, 256, 0, stream>>>(W_ih, W_hh, Whm);

  // Conv blocks (WMMA GEMM + fused LayerNorm/ReLU/mask)
  dim3 cgrid(Ssz / 16, Bsz);
  conv_ln_relu_kernel<float><<<cgrid, 256, 0, stream>>>(
      h_text, conv_b1, ln_g1, ln_b1, mask, Wc1, x1);
  conv_ln_relu_kernel<_Float16><<<cgrid, 256, 0, stream>>>(
      x1, conv_b2, ln_g2, ln_b2, mask, Wc2, x2);

  // Hoisted input projection for all timesteps (parallel WMMA GEMM)
  dim3 xgrid((Bsz * Ssz) / 16, G3 / 512);
  xproj_kernel<<<xgrid, 256, 0, stream>>>(x2, Wxp, b_ih, Xproj);

  // Persistent GRU recurrence (grid barrier per step, WMMA per step)
  gru_kernel<<<64, 256, 0, stream>>>(Xproj, Whm, b_hh, mask, h16, h32, outs, bar);

  // Bottleneck projection
  bottleneck_kernel<<<(Bsz * Ssz) / 8, 256, 0, stream>>>(outs, W_bn, b_bn, out);
}
